// TrainModel_72851235274861
// MI455X (gfx1250) — matrix operand order repaired and checked by reading, not compiled, not run on hardware
//
#include <hip/hip_runtime.h>
#include <hip/hip_fp16.h>

// ---------------------------------------------------------------------------
// CNN + BiLSTM encoder + attention decoder, CDNA5 (gfx1250) WMMA version.
//
// GEMMs use a wave32 WMMA kernel (v_wmma_f32_16x16x32_f16), one wave per
// 16x64 output block (1 A-frag reused across 4 B-frags / 4 WMMAs per K-chunk):
//   - A operand: row-major f16, per-lane frag = two global b128 loads
//     (ISA 16-bit A 16x32 layout: halves 0..7 -> K=kb..kb+7,
//      halves 8..15 -> K=kb+16..kb+23, kb = lane<16 ? 0 : 8)
//   - B operand: weights pre-transposed to (N x Kpad) f16, per-lane frag =
//     two global b128 loads (lanes 0..15: N=lane, K=0..7,16..23; lanes 16..31: K=8..15,24..31)
//   - C/D: v8f, row = m0 + r + (lane<16?0:8), col = n0 + (lane&15)
// Weights all fit in the 192MB L2, so per-step re-streaming of W is cheap;
// the network is latency-bound on the sequential LSTM/decoder scans, so we
// keep those as compact per-step kernel chains (graph-captured once).
// The attention kernel stages its broadcast vectors into LDS with
// GLOBAL_LOAD_ASYNC_TO_LDS_B32 + s_wait_asynccnt (builtin confirmed present;
// pointer params are int* in AS1/AS3 per the compiler diagnostic).
// ---------------------------------------------------------------------------

typedef __attribute__((ext_vector_type(16))) _Float16 v16h;
typedef __attribute__((ext_vector_type(8)))  _Float16 v8h;
typedef __attribute__((ext_vector_type(8)))  float    v8f;

// ----------------------------- utility kernels -----------------------------

__global__ void k_zero_u32(unsigned int* __restrict__ p, long n) {
  long i = (long)blockIdx.x * blockDim.x + threadIdx.x;
  if (i < n) p[i] = 0u;
}

// W (K x N, f32, row-major)  ->  Bt (Npad x Kpad, f16), zero-padded.
__global__ void k_cvt_transpose(const float* __restrict__ W, _Float16* __restrict__ Bt,
                                int K, int N, int Kpad, int Npad) {
  long idx = (long)blockIdx.x * blockDim.x + threadIdx.x;
  long total = (long)Npad * Kpad;
  if (idx >= total) return;
  int k = (int)(idx % Kpad);
  int n = (int)(idx / Kpad);
  float v = (k < K && n < N) ? W[(long)k * N + n] : 0.f;
  Bt[idx] = (_Float16)v;
}

// im2col for 3x3 SAME conv, NHWC f32 -> (B*H*W, Kpad) f16, zero padded.
__global__ void k_im2col(const float* __restrict__ src, _Float16* __restrict__ A,
                         int Bn, int H, int W, int C, int Kpad) {
  long idx = (long)blockIdx.x * blockDim.x + threadIdx.x;
  long total = (long)Bn * H * W * Kpad;
  if (idx >= total) return;
  int k = (int)(idx % Kpad);
  long m = idx / Kpad;
  int x = (int)(m % W); long t = m / W;
  int y = (int)(t % H); int b = (int)(t / H);
  _Float16 val = (_Float16)0.f;
  int K = 9 * C;
  if (k < K) {
    int ci = k % C; int r = k / C;
    int kx = r % 3, ky = r / 3;
    int yy = y + ky - 1, xx = x + kx - 1;
    if (yy >= 0 && yy < H && xx >= 0 && xx < W)
      val = (_Float16)src[(((long)b * H + yy) * W + xx) * C + ci];
  }
  A[idx] = val;
}

// 2x2 max pool, stride 2, NHWC f32.
__global__ void k_maxpool(const float* __restrict__ src, float* __restrict__ dst,
                          int Bn, int H, int W, int C) {
  int Ho = H >> 1, Wo = W >> 1;
  long idx = (long)blockIdx.x * blockDim.x + threadIdx.x;
  long total = (long)Bn * Ho * Wo * C;
  if (idx >= total) return;
  int c = (int)(idx % C); long t = idx / C;
  int x = (int)(t % Wo); t /= Wo;
  int y = (int)(t % Ho); int b = (int)(t / Ho);
  const float* p = src + (((long)b * H + 2 * y) * W + 2 * x) * C + c;
  long rs = (long)W * C;
  float m = fmaxf(fmaxf(p[0], p[C]), fmaxf(p[rs], p[rs + C]));
  dst[idx] = m;
}

// ------------------------------- WMMA GEMM ---------------------------------

// A frag: p = row_base + kc ; halves 0..7 at +kbA, halves 8..15 at +kbA+16.
__device__ __forceinline__ v16h load_a_frag(const _Float16* p) {
  const v8h lo = *(const v8h*)p;
  const v8h hi = *(const v8h*)(p + 16);
  v16h r;
  #pragma unroll
  for (int i = 0; i < 8; ++i) { r[i] = lo[i]; r[i + 8] = hi[i]; }
  return r;
}
// B frag: p = row_base + kc + kbB ; halves 0..7 at +kbB, halves 8..15 at +kbB+16.
__device__ __forceinline__ v16h load_b_frag(const _Float16* p) {
  const v8h lo = *(const v8h*)p;
  const v8h hi = *(const v8h*)(p + 16);
  v16h r;
  #pragma unroll
  for (int i = 0; i < 8; ++i) { r[i] = lo[i]; r[i + 8] = hi[i]; }
  return r;
}

// C(M,N) f32 = act( A(M,K) f16 * Bt(N,K)^T f16 + bias ), optional f16 mirror.
// One wave computes a 16x64 block (4 accumulators, shared A frag);
// 8 waves / 256-thread block (wave32). N must be a multiple of 64.
__global__ void k_gemm_wmma(const _Float16* __restrict__ A, int lda,
                            const _Float16* __restrict__ Bt, int ldb,
                            float* __restrict__ C, long ldc,
                            _Float16* __restrict__ C16, int ldc16,
                            const float* __restrict__ bias,
                            int M, int N, int K, int Nreal, int act) {
  const int lane = threadIdx.x & 31;
  const int wave = threadIdx.x >> 5;
  const int ntn = N >> 6;                       // 64-wide N blocks
  const int tile = blockIdx.x * 8 + wave;       // wave-uniform
  if (tile >= (M >> 4) * ntn) return;           // whole wave exits together
  const int m0 = (tile / ntn) << 4;
  const int n0 = (tile % ntn) << 6;

  const int l15 = lane & 15;
  const int kbA = (lane < 16) ? 0 : 8;          // A frag K base per half-wave
  const int kbB = (lane < 16) ? 0 : 8;          // B frag K base per half-wave
  const _Float16* aptr = A + (long)(m0 + l15) * lda + kbA;
  const _Float16* bp0  = Bt + (long)(n0 + l15) * ldb + kbB;
  const _Float16* bp1  = bp0 + (long)16 * ldb;
  const _Float16* bp2  = bp0 + (long)32 * ldb;
  const _Float16* bp3  = bp0 + (long)48 * ldb;

  v8f acc0 = {}, acc1 = {}, acc2 = {}, acc3 = {};
  for (int kc = 0; kc < K; kc += 32) {
    if (kc + 32 < K) {                          // pull next chunk into near $
      __builtin_prefetch(aptr + kc + 32, 0, 3);
      __builtin_prefetch(bp0 + kc + 32, 0, 3);
      __builtin_prefetch(bp1 + kc + 32, 0, 3);
      __builtin_prefetch(bp2 + kc + 32, 0, 3);
      __builtin_prefetch(bp3 + kc + 32, 0, 3);
    }
    const v16h a  = load_a_frag(aptr + kc);
    const v16h b0 = load_b_frag(bp0 + kc);
    const v16h b1 = load_b_frag(bp1 + kc);
    const v16h b2 = load_b_frag(bp2 + kc);
    const v16h b3 = load_b_frag(bp3 + kc);
    acc0 = __builtin_amdgcn_wmma_f32_16x16x32_f16(false, a, false, b0, (short)0, acc0, false, false);
    acc1 = __builtin_amdgcn_wmma_f32_16x16x32_f16(false, a, false, b1, (short)0, acc1, false, false);
    acc2 = __builtin_amdgcn_wmma_f32_16x16x32_f16(false, a, false, b2, (short)0, acc2, false, false);
    acc3 = __builtin_amdgcn_wmma_f32_16x16x32_f16(false, a, false, b3, (short)0, acc3, false, false);
  }

  #pragma unroll
  for (int j = 0; j < 4; ++j) {
    const v8f acc = (j == 0) ? acc0 : (j == 1) ? acc1 : (j == 2) ? acc2 : acc3;
    const int col = n0 + 16 * j + l15;
    if (col < Nreal) {
      const float bv = bias ? bias[col] : 0.f;
      #pragma unroll
      for (int r = 0; r < 8; ++r) {
        int row = m0 + r + ((lane < 16) ? 0 : 8);
        float vz = acc[r] + bv;
        if (act == 1) vz = vz > 0.f ? vz : 0.f;
        if (C)   C[(long)row * ldc + col] = vz;
        if (C16) C16[(long)row * ldc16 + col] = (_Float16)vz;
      }
    }
  }
}

// ------------------------------ LSTM gates ---------------------------------
// z (M,4H): [i | j | f | o].  c updated in place; h written f32 (+ up to two
// f16 mirrors, used to build the next step's concatenated GEMM A operand).
__global__ void k_lstm_gates(const float* __restrict__ z, float* __restrict__ c,
                             float* __restrict__ hout,
                             _Float16* h16a, int lda16,
                             _Float16* h16b, int ldb16,
                             int M, int Hh) {
  int idx = blockIdx.x * blockDim.x + threadIdx.x;
  if (idx >= M * Hh) return;
  int m = idx / Hh, d = idx % Hh;
  const float* zr = z + (long)m * 4 * Hh;
  float gi = zr[d], gj = zr[Hh + d], gf = zr[2 * Hh + d], go = zr[3 * Hh + d];
  float sf = 1.f / (1.f + __expf(-(gf + 1.f)));
  float si = 1.f / (1.f + __expf(-gi));
  float cn = sf * c[idx] + si * tanhf(gj);
  c[idx] = cn;
  float h = (1.f / (1.f + __expf(-go))) * tanhf(cn);
  hout[idx] = h;
  if (h16a) h16a[(long)m * lda16 + d] = (_Float16)h;
  if (h16b) h16b[(long)m * ldb16 + d] = (_Float16)h;
}

// pool4 (16,8,64,512) -> x_cat_fw/bw[t][n][0:512] f16 (t = wp, n = b*8+hp).
__global__ void k_build_enc_inputs(const float* __restrict__ pool4,
                                   _Float16* __restrict__ xfw,
                                   _Float16* __restrict__ xbw) {
  long idx = (long)blockIdx.x * blockDim.x + threadIdx.x;
  if (idx >= (long)64 * 128 * 512) return;
  int cdim = (int)(idx % 512);
  int n = (int)((idx / 512) % 128);
  int t = (int)(idx / (512 * 128));
  _Float16 v = (_Float16)pool4[((long)n * 64 + t) * 512 + cdim];
  xfw[((long)t * 128 + n) * 768 + cdim] = v;
  xbw[((long)(63 - t) * 128 + n) * 768 + cdim] = v;
}

// memory (16,512,512) f32 + f16 mirror from hs_fw / hs_bw(step-indexed).
__global__ void k_build_memory(const float* __restrict__ hsfw,
                               const float* __restrict__ hsbw,
                               float* __restrict__ mem,
                               _Float16* __restrict__ mem16) {
  long idx = (long)blockIdx.x * blockDim.x + threadIdx.x;
  if (idx >= (long)16 * 512 * 512) return;
  int d = (int)(idx % 512);
  int l = (int)((idx / 512) % 512);
  int b = (int)(idx / (512 * 512));
  int t = l % 64, hp = l / 64;
  int n = b * 8 + hp;
  float v = (d < 256) ? hsfw[((long)t * 128 + n) * 256 + d]
                      : hsbw[((long)(63 - t) * 128 + n) * 256 + (d - 256)];
  mem[idx] = v;
  mem16[idx] = (_Float16)v;
}

// embedding gather: out[t][b][e] = E[labels[b][t]][e], t in [0,150)
__global__ void k_embed(const float* __restrict__ E, const int* __restrict__ labels,
                        _Float16* __restrict__ out) {
  int idx = blockIdx.x * blockDim.x + threadIdx.x;
  if (idx >= 150 * 16 * 80) return;
  int e = idx % 80; int tb = idx / 80;
  int b = tb % 16; int t = tb / 16;
  out[idx] = (_Float16)E[(long)labels[b * 151 + t] * 80 + e];
}

__global__ void k_copy_emb(const _Float16* __restrict__ src, _Float16* __restrict__ xcat) {
  int idx = blockIdx.x * blockDim.x + threadIdx.x;
  if (idx >= 16 * 80) return;
  int e = idx % 80, b = idx / 80;
  xcat[b * 1120 + e] = src[idx];
}

// Fused attention: score = tanh(keys[b]+q[b])·v ; softmax over L=512 ;
// ctx = align @ memory[b] -> f16 into hc[:,512:1024]. One block (512T) per b.
// q row + v_att staged into LDS via async-load-to-LDS when available.
__global__ void k_attention(const float* __restrict__ q, const float* __restrict__ keys,
                            const float* __restrict__ mem, const float* __restrict__ vatt,
                            _Float16* __restrict__ ctx16 /* = hc + 512 */) {
  const int b = blockIdx.x;
  const int tid = threadIdx.x;   // 512 threads
  __shared__ float sQ[512];
  __shared__ float sV[512];
  __shared__ float sAl[512];
  __shared__ float sTmp[512];
  const float* qb = q + (long)b * 512;

#if __has_builtin(__builtin_amdgcn_global_load_async_to_lds_b32)
  // ASYNCcnt path: each lane asynchronously copies one dword of q[b] and
  // v_att into LDS (GLOBAL_LOAD_ASYNC_TO_LDS_B32), then waits on ASYNCcnt.
  // Builtin types its pointers as int* (AS1 src, AS3 dst).
  __builtin_amdgcn_global_load_async_to_lds_b32(
      (__attribute__((address_space(1))) int*)(qb + tid),
      (__attribute__((address_space(3))) int*)(&sQ[tid]), 0, 0);
  __builtin_amdgcn_global_load_async_to_lds_b32(
      (__attribute__((address_space(1))) int*)(vatt + tid),
      (__attribute__((address_space(3))) int*)(&sV[tid]), 0, 0);
#if __has_builtin(__builtin_amdgcn_s_wait_asynccnt)
  __builtin_amdgcn_s_wait_asynccnt(0);
#else
  asm volatile("s_wait_asynccnt 0" ::: "memory");
#endif
#else
  sQ[tid] = qb[tid];
  sV[tid] = vatt[tid];
#endif
  __syncthreads();

  const float* kb = keys + ((long)b * 512 + tid) * 512;
  float s = 0.f;
  #pragma unroll 4
  for (int d = 0; d < 512; ++d) s += tanhf(kb[d] + sQ[d]) * sV[d];
  sAl[tid] = s; sTmp[tid] = s;
  __syncthreads();
  for (int off = 256; off > 0; off >>= 1) {
    if (tid < off) sTmp[tid] = fmaxf(sTmp[tid], sTmp[tid + off]);
    __syncthreads();
  }
  const float mx = sTmp[0];
  __syncthreads();
  const float e = __expf(sAl[tid] - mx);
  sAl[tid] = e; sTmp[tid] = e;
  __syncthreads();
  for (int off = 256; off > 0; off >>= 1) {
    if (tid < off) sTmp[tid] += sTmp[tid + off];
    __syncthreads();
  }
  const float inv = 1.f / sTmp[0];
  const float* mb = mem + (long)b * 512 * 512;
  float acc = 0.f;
  for (int l = 0; l < 512; ++l) acc += sAl[l] * mb[(long)l * 512 + tid];
  ctx16[(long)b * 1024 + tid] = (_Float16)(acc * inv);
}

// --------------------------------- driver ----------------------------------

extern "C" void kernel_launch(void* const* d_in, const int* in_sizes, int n_in,
                              void* d_out, int out_size, void* d_ws, size_t ws_size,
                              hipStream_t stream) {
  (void)in_sizes; (void)n_in; (void)out_size; (void)ws_size;
  const float* inp   = (const float*)d_in[0];
  const int*   labs  = (const int*)  d_in[1];
  // d_in[2] = num_words (151) — T_dec=150 baked in, matching the reference.
  const float* ck1 = (const float*)d_in[3],  *cb1 = (const float*)d_in[4];
  const float* ck2 = (const float*)d_in[5],  *cb2 = (const float*)d_in[6];
  const float* ck3 = (const float*)d_in[7],  *cb3 = (const float*)d_in[8];
  const float* ck4 = (const float*)d_in[9],  *cb4 = (const float*)d_in[10];
  const float* Wfw = (const float*)d_in[11], *bfw = (const float*)d_in[12];
  const float* Wbw = (const float*)d_in[13], *bbw = (const float*)d_in[14];
  const float* Wdec= (const float*)d_in[15], *bdec= (const float*)d_in[16];
  const float* Wmem= (const float*)d_in[17];
  const float* Wq  = (const float*)d_in[18];
  const float* vat = (const float*)d_in[19];
  const float* Wat = (const float*)d_in[20];
  const float* Wout= (const float*)d_in[21], *bout= (const float*)d_in[22];
  const float* Emb = (const float*)d_in[23];
  float* out = (float*)d_out;

  // ---- workspace bump allocator (regions reused across conv layers) ----
  char* ws = (char*)d_ws;
  size_t off = 0;
  auto alloc = [&](size_t bytes) -> char* {
    char* p = ws + off; off += (bytes + 255) & ~(size_t)255; return p;
  };
  char* bigA  = alloc((size_t)32768 * 2304 * 2);   // largest im2col (A2/A4)
  char* bigO  = alloc((size_t)524288 * 64 * 4);    // largest conv out (out1)
  char* pool1 = alloc((size_t)16 * 32 * 256 * 64 * 4);
  char* pool2 = alloc((size_t)16 * 16 * 128 * 128 * 4);
  char* pool4 = alloc((size_t)16 * 8 * 64 * 512 * 4);
  char* w1t   = alloc((size_t)64 * 32 * 2);
  char* w2t   = alloc((size_t)128 * 576 * 2);
  char* w3t   = alloc((size_t)256 * 1152 * 2);
  char* w4t   = alloc((size_t)512 * 2304 * 2);
  char* wfwT  = alloc((size_t)1024 * 768 * 2);
  char* wbwT  = alloc((size_t)1024 * 768 * 2);
  char* wdecT = alloc((size_t)2048 * 1120 * 2);
  char* wmemT = alloc((size_t)512 * 512 * 2);
  char* wqT   = alloc((size_t)512 * 512 * 2);
  char* watT  = alloc((size_t)512 * 1024 * 2);
  char* woutT = alloc((size_t)512 * 512 * 2);     // N padded 504 -> 512
  char* xfw   = alloc((size_t)64 * 128 * 768 * 2);
  char* xbw   = alloc((size_t)64 * 128 * 768 * 2);
  char* zenc  = alloc((size_t)128 * 1024 * 4);
  char* cfw   = alloc((size_t)128 * 256 * 4);
  char* cbw   = alloc((size_t)128 * 256 * 4);
  char* hsfw  = alloc((size_t)64 * 128 * 256 * 4);
  char* hsbw  = alloc((size_t)64 * 128 * 256 * 4);
  char* memf  = alloc((size_t)16 * 512 * 512 * 4);
  char* mem16 = alloc((size_t)16 * 512 * 512 * 2);
  char* keys  = alloc((size_t)16 * 512 * 512 * 4);
  char* emb16 = alloc((size_t)150 * 16 * 80 * 2);
  char* xdec  = alloc((size_t)16 * 1120 * 2);
  char* hcbuf = alloc((size_t)16 * 1024 * 2);
  char* zdec  = alloc((size_t)16 * 2048 * 4);
  char* cdec  = alloc((size_t)16 * 512 * 4);
  char* hdec  = alloc((size_t)16 * 512 * 4);
  char* qbuf  = alloc((size_t)16 * 512 * 4);

  auto cdiv = [](long a, long b) { return (a + b - 1) / b; };
  auto zero = [&](void* p, size_t bytes) {
    long n = (long)(bytes / 4);
    k_zero_u32<<<dim3((unsigned)cdiv(n, 256)), dim3(256), 0, stream>>>((unsigned int*)p, n);
  };
  auto cvtT = [&](const float* W, void* Bt, int K, int N, int Kpad, int Npad) {
    long n = (long)Npad * Kpad;
    k_cvt_transpose<<<dim3((unsigned)cdiv(n, 256)), dim3(256), 0, stream>>>(
        W, (_Float16*)Bt, K, N, Kpad, Npad);
  };
  auto gemm = [&](const void* A, int lda, const void* Bt, int ldb,
                  void* C, long ldc, void* C16, int ldc16, const float* bias,
                  int M, int N, int K, int Nreal, int act) {
    int tiles = (M / 16) * (N / 64);   // one wave per 16x64 block
    k_gemm_wmma<<<dim3((unsigned)cdiv(tiles, 8)), dim3(256), 0, stream>>>(
        (const _Float16*)A, lda, (const _Float16*)Bt, ldb,
        (float*)C, ldc, (_Float16*)C16, ldc16, bias, M, N, K, Nreal, act);
  };

  // ---- weight prep (f32 -> transposed padded f16) ----
  cvtT(ck1,  w1t,    9,   64,   32,   64);
  cvtT(ck2,  w2t,  576,  128,  576,  128);
  cvtT(ck3,  w3t, 1152,  256, 1152,  256);
  cvtT(ck4,  w4t, 2304,  512, 2304,  512);
  cvtT(Wfw,  wfwT, 768, 1024,  768, 1024);
  cvtT(Wbw,  wbwT, 768, 1024,  768, 1024);
  cvtT(Wdec, wdecT,1104,2048, 1120, 2048);
  cvtT(Wmem, wmemT, 512,  512,  512,  512);
  cvtT(Wq,   wqT,   512,  512,  512,  512);
  cvtT(Wat,  watT, 1024,  512, 1024,  512);
  cvtT(Wout, woutT, 512,  504,  512,  512);

  // ---- conv stack (im2col + WMMA GEMM, bias+ReLU fused) ----
  auto im2col = [&](const void* src, void* A, int Bn, int H, int W, int C, int Kpad) {
    long n = (long)Bn * H * W * Kpad;
    k_im2col<<<dim3((unsigned)cdiv(n, 256)), dim3(256), 0, stream>>>(
        (const float*)src, (_Float16*)A, Bn, H, W, C, Kpad);
  };
  auto pool = [&](const void* src, void* dst, int Bn, int H, int W, int C) {
    long n = (long)Bn * (H / 2) * (W / 2) * C;
    k_maxpool<<<dim3((unsigned)cdiv(n, 256)), dim3(256), 0, stream>>>(
        (const float*)src, (float*)dst, Bn, H, W, C);
  };
  im2col(inp,   bigA, 16, 64, 512,   1,   32);
  gemm(bigA,   32, w1t,   32, bigO,   64, nullptr, 0, cb1, 524288,  64,   32,  64, 1);
  pool(bigO, pool1, 16, 64, 512, 64);
  im2col(pool1, bigA, 16, 32, 256,  64,  576);
  gemm(bigA,  576, w2t,  576, bigO,  128, nullptr, 0, cb2, 131072, 128,  576, 128, 1);
  pool(bigO, pool2, 16, 32, 256, 128);
  im2col(pool2, bigA, 16, 16, 128, 128, 1152);
  gemm(bigA, 1152, w3t, 1152, bigO,  256, nullptr, 0, cb3,  32768, 256, 1152, 256, 1);
  im2col(bigO,  bigA, 16, 16, 128, 256, 2304);   // conv3 out consumed here
  gemm(bigA, 2304, w4t, 2304, bigO,  512, nullptr, 0, cb4,  32768, 512, 2304, 512, 1);
  pool(bigO, pool4, 16, 16, 128, 512);

  // ---- bidirectional LSTM encoder (64 steps, batch 128, hidden 256) ----
  zero(xfw, (size_t)64 * 128 * 768 * 2);          // h-slots + t=0 start at 0
  zero(xbw, (size_t)64 * 128 * 768 * 2);
  zero(cfw, (size_t)128 * 256 * 4);
  zero(cbw, (size_t)128 * 256 * 4);
  {
    long n = (long)64 * 128 * 512;
    k_build_enc_inputs<<<dim3((unsigned)cdiv(n, 256)), dim3(256), 0, stream>>>(
        (const float*)pool4, (_Float16*)xfw, (_Float16*)xbw);
  }
  for (int t = 0; t < 64; ++t) {
    // forward
    gemm(xfw + (size_t)t * 128 * 768 * 2, 768, wfwT, 768,
         zenc, 1024, nullptr, 0, bfw, 128, 1024, 768, 1024, 0);
    _Float16* nfw = (t < 63) ? (_Float16*)(xfw + (size_t)(t + 1) * 128 * 768 * 2) + 512
                             : nullptr;
    k_lstm_gates<<<dim3(128), dim3(256), 0, stream>>>(
        (const float*)zenc, (float*)cfw,
        (float*)(hsfw + (size_t)t * 128 * 256 * 4),
        nfw, 768, (_Float16*)nullptr, 0, 128, 256);
    // backward (step-indexed; inputs already reversed in xbw)
    gemm(xbw + (size_t)t * 128 * 768 * 2, 768, wbwT, 768,
         zenc, 1024, nullptr, 0, bbw, 128, 1024, 768, 1024, 0);
    _Float16* nbw = (t < 63) ? (_Float16*)(xbw + (size_t)(t + 1) * 128 * 768 * 2) + 512
                             : nullptr;
    k_lstm_gates<<<dim3(128), dim3(256), 0, stream>>>(
        (const float*)zenc, (float*)cbw,
        (float*)(hsbw + (size_t)t * 128 * 256 * 4),
        nbw, 768, (_Float16*)nullptr, 0, 128, 256);
  }
  {
    long n = (long)16 * 512 * 512;
    k_build_memory<<<dim3((unsigned)cdiv(n, 256)), dim3(256), 0, stream>>>(
        (const float*)hsfw, (const float*)hsbw, (float*)memf, (_Float16*)mem16);
  }
  // keys = memory @ Wmem   (8192 x 512 x 512)
  gemm(mem16, 512, wmemT, 512, keys, 512, nullptr, 0, nullptr, 8192, 512, 512, 512, 0);

  // ---- attention decoder (150 steps, batch 16) ----
  k_embed<<<dim3((unsigned)cdiv(150 * 16 * 80, 256)), dim3(256), 0, stream>>>(
      Emb, labs, (_Float16*)emb16);
  zero(xdec, (size_t)16 * 1120 * 2);   // attn0 = h0 = 0, pad cols stay 0
  zero(hcbuf,(size_t)16 * 1024 * 2);
  zero(cdec, (size_t)16 * 512 * 4);
  for (int t = 0; t < 150; ++t) {
    // x_cat = [emb_t | attn | h] (f16, K padded 1104 -> 1120)
    k_copy_emb<<<dim3(5), dim3(256), 0, stream>>>(
        (const _Float16*)(emb16 + (size_t)t * 16 * 80 * 2), (_Float16*)xdec);
    // z = x_cat @ Wdec + bdec     (16 x 1120 x 2048)
    gemm(xdec, 1120, wdecT, 1120, zdec, 2048, nullptr, 0, bdec, 16, 2048, 1120, 2048, 0);
    // gates: h -> f16 into x_cat[:,592:1104] and hc[:,0:512]
    k_lstm_gates<<<dim3(32), dim3(256), 0, stream>>>(
        (const float*)zdec, (float*)cdec, (float*)hdec,
        (_Float16*)xdec + 592, 1120, (_Float16*)hcbuf, 1024, 16, 512);
    // q = h @ Wq                  (16 x 512 x 512), A = hc[:,0:512]
    gemm(hcbuf, 1024, wqT, 512, qbuf, 512, nullptr, 0, nullptr, 16, 512, 512, 512, 0);
    // attention -> ctx f16 into hc[:,512:1024]
    k_attention<<<dim3(16), dim3(512), 0, stream>>>(
        (const float*)qbuf, (const float*)keys, (const float*)memf, vat,
        (_Float16*)hcbuf + 512);
    // attn = [h|ctx] @ Wattn -> f16 straight into x_cat[:,80:592]
    gemm(hcbuf, 1024, watT, 1024, nullptr, 0, (_Float16*)xdec + 80, 1120,
         nullptr, 16, 512, 1024, 512, 0);
    // logits = attn @ Wout + bout -> d_out[:, t, 0:504]  (N padded to 512)
    gemm((_Float16*)xdec + 80, 1120, woutT, 512,
         out + (size_t)t * 504, (long)150 * 504, nullptr, 0, bout,
         16, 512, 512, 504, 0);
  }
}
